// DARTSCell_47021301957139
// MI455X (gfx1250) — matrix operand order repaired and checked by reading, not compiled, not run on hardware
//
#include <hip/hip_runtime.h>
#include <math.h>

static constexpr int NH  = 850;    // NHID == NINP
static constexpr int NX  = 1700;   // 2*NHID
static constexpr int NHP = 896;    // padded to multiple of 128
static constexpr int NXP = 1792;   // padded to multiple of 128
static constexpr int B_  = 256;
static constexpr int T_  = 128;

typedef __attribute__((ext_vector_type(16))) __bf16 v16bf;
typedef __attribute__((ext_vector_type(8)))  float  v8f;

union Frag16 { v16bf v; uint4 q[2]; };

__device__ __forceinline__ unsigned short f2bf(float f) {
  unsigned int u = __float_as_uint(f);
  u += 0x7FFFu + ((u >> 16) & 1u);          // round-to-nearest-even
  return (unsigned short)(u >> 16);
}

__device__ __forceinline__ float sigmoidf_(float x) { return 1.0f / (1.0f + __expf(-x)); }

// ---- CDNA5 async global->LDS copy (ASYNCcnt-tracked, no VGPR transit) ----
// GV mode: LDS address comes from the vdst VGPR; global addr = 64-bit VGPR pair.
__device__ __forceinline__ void async_cp_b128(const unsigned short* g, unsigned short* l) {
  unsigned lds32 = (unsigned)(size_t)l;                 // generic->LDS offset (low 32 bits)
  unsigned long long ga = (unsigned long long)(size_t)g;
  asm volatile("global_load_async_to_lds_b128 %0, %1, off"
               :: "v"(lds32), "v"(ga) : "memory");
}

template <int N>
__device__ __forceinline__ void wait_async_le() {
#if __has_builtin(__builtin_amdgcn_s_wait_asynccnt)
  __builtin_amdgcn_s_wait_asynccnt(N);
#else
  if (N == 0) asm volatile("s_wait_asynccnt 0x0" ::: "memory");
  else        asm volatile("s_wait_asynccnt 0x8" ::: "memory");
#endif
}

// ---------------------------------------------------------------------------
// prep / elementwise kernels
// ---------------------------------------------------------------------------

__global__ void k_zero32(float* __restrict__ p, int n) {
  int i = blockIdx.x * blockDim.x + threadIdx.x;
  if (i < n) p[i] = 0.0f;
}

// dst[k*NP + n] = bf16(src[k*lds + n]) with zero padding (row-major K x N kept)
__global__ void k_convpad_bf16(const float* __restrict__ src, unsigned short* __restrict__ dst,
                               int Ks, int Ns, int lds_, int KP, int NP) {
  int idx = blockIdx.x * blockDim.x + threadIdx.x;
  if (idx >= KP * NP) return;
  int k = idx / NP, n = idx - k * NP;
  dst[idx] = (k < Ks && n < Ns) ? f2bf(src[(size_t)k * lds_ + n]) : (unsigned short)0;
}

__global__ void k_sigpad(const float* __restrict__ src, float* __restrict__ dst, int n, int np) {
  int i = blockIdx.x * blockDim.x + threadIdx.x;
  if (i < np) dst[i] = (i < n) ? src[i] : 0.0f;
}

__global__ void k_inith(const float* __restrict__ h0, float* __restrict__ hf) {
  int i = blockIdx.x * blockDim.x + threadIdx.x;
  if (i >= B_ * NHP) return;
  int r = i / NHP, c = i - r * NHP;
  hf[i] = (c < NH) ? h0[r * NH + c] : 0.0f;
}

// xh_bf16[B, NXP] = [x_t | h | 0-pad], converted to bf16
__global__ void k_packxh(const float* __restrict__ xt, const float* __restrict__ hf,
                         unsigned short* __restrict__ xh) {
  int i = blockIdx.x * blockDim.x + threadIdx.x;
  if (i >= B_ * NXP) return;
  int r = i / NXP, c = i - r * NXP;
  float v = 0.0f;
  if (c < NH)      v = xt[r * NH + c];
  else if (c < NX) v = hf[r * NHP + (c - NH)];
  xh[i] = f2bf(v);
}

// s0 = h + sigmoid(ch[:, :NH]) * (tanh(ch[:, NH:]) - h)
__global__ void k_epi0(const float* __restrict__ ch, const float* __restrict__ hf,
                       float* __restrict__ sf, unsigned short* __restrict__ sbf) {
  int i = blockIdx.x * blockDim.x + threadIdx.x;
  if (i >= B_ * NH) return;
  int r = i / NH, c = i - r * NH;
  float c0 = sigmoidf_(ch[r * NXP + c]);
  float h0 = tanhf(ch[r * NXP + c + NH]);
  float hp = hf[r * NHP + c];
  float s  = hp + c0 * (h0 - hp);
  sf[r * NHP + c]  = s;
  sbf[r * NHP + c] = f2bf(s);
}

// s_new = sp + sigmoid(ch[:, :NH]) * (act(ch[:, NH:]) - sp); act: 0=tanh 1=relu 2=sigmoid 3=id
__global__ void k_epi(const float* __restrict__ ch, const float* __restrict__ spf,
                      float* __restrict__ sf, unsigned short* __restrict__ sbf, int act) {
  int i = blockIdx.x * blockDim.x + threadIdx.x;
  if (i >= B_ * NH) return;
  int r = i / NH, c = i - r * NH;
  float cg = sigmoidf_(ch[r * NXP + c]);
  float v  = ch[r * NXP + c + NH];
  float hh;
  if      (act == 0) hh = tanhf(v);
  else if (act == 1) hh = fmaxf(v, 0.0f);
  else if (act == 2) hh = sigmoidf_(v);
  else               hh = v;
  float sp = spf[r * NHP + c];
  float s  = sp + cg * (hh - sp);
  sf[r * NHP + c]  = s;
  sbf[r * NHP + c] = f2bf(s);
}

// h_new = mean(states[1..8]); writes output slice (+ optional final-state copy) and next h
__global__ void k_final(const float* __restrict__ stf, float* __restrict__ out_t,
                        float* __restrict__ hf, float* __restrict__ out_last) {
  int i = blockIdx.x * blockDim.x + threadIdx.x;
  if (i >= B_ * NH) return;
  int r = i / NH, c = i - r * NH;
  float s = 0.0f;
#pragma unroll
  for (int k = 1; k <= 8; ++k) s += stf[(size_t)k * B_ * NHP + r * NHP + c];
  s *= 0.125f;
  out_t[i] = s;
  hf[r * NHP + c] = s;
  if (out_last) out_last[i] = s;
}

// ---------------------------------------------------------------------------
// bf16 WMMA GEMM:  D[M,N] = A[M,K] * Bw[K,N]   (M = gridDim.y*128 = 256)
//   sigma != null : out bf16 = acc * sigma[n]   (fused diag scale for low-rank)
//   sigma == null : out f32  = acc
// 256 threads = 8 waves in 2(M) x 4(N); wave tile 64x32 = 4x2 WMMA accumulators.
// Double-buffered LDS tiles filled with global_load_async_to_lds_b128 so the
// fetch of chunk j+1 overlaps WMMA compute on chunk j (ASYNCcnt pipeline).
// ---------------------------------------------------------------------------
__global__ __launch_bounds__(256)
void k_gemm_bf16(const unsigned short* __restrict__ A,
                 const unsigned short* __restrict__ Bw,
                 int K, int N,
                 const float* __restrict__ sigma,
                 unsigned short* __restrict__ outB,
                 float* __restrict__ outF) {
  __shared__ unsigned short As[2][128][72];    // 128 rows x 64 K  (+8 pad)
  __shared__ unsigned short Bs[2][64][136];    // 64 K x 128 cols  (+8 pad)

  const int tid   = threadIdx.x;
  const int lane  = tid & 31;
  const int wave  = tid >> 5;
  const int wm    = wave >> 2;              // 0..1  (M)
  const int wn    = wave & 3;               // 0..3  (N)
  const int mBase = blockIdx.y * 128;
  const int nBase = blockIdx.x * 128;
  const int mrow  = lane & 15;
  const int khalf = lane >> 4;

  v8f acc[4][2];
#pragma unroll
  for (int i = 0; i < 4; ++i)
#pragma unroll
    for (int j = 0; j < 2; ++j)
#pragma unroll
      for (int e = 0; e < 8; ++e) acc[i][j][e] = 0.0f;

  auto issue_chunk = [&](int kc, int buf) {
#pragma unroll
    for (int i = 0; i < 4; ++i) {
      int idx = i * 256 + tid;
      int ra = idx >> 3, ca = (idx & 7) << 3;     // A: 128 rows x 8 chunks
      async_cp_b128(&A[(size_t)(mBase + ra) * K + kc + ca], &As[buf][ra][ca]);
      int rb = idx >> 4, cb = (idx & 15) << 3;    // B: 64 rows x 16 chunks
      async_cp_b128(&Bw[(size_t)(kc + rb) * N + nBase + cb], &Bs[buf][rb][cb]);
    }
  };

  const int nc = K >> 6;                    // 64-wide K chunks
  issue_chunk(0, 0);

  for (int j = 0; j < nc; ++j) {
    const int buf = j & 1;
    if (j + 1 < nc) {
      issue_chunk((j + 1) << 6, buf ^ 1);   // prefetch next chunk into other buffer
      wait_async_le<8>();                   // newest 8 = next chunk; chunk j has landed
    } else {
      wait_async_le<0>();
    }
    __syncthreads();                        // everyone's chunk-j tiles visible

#pragma unroll
    for (int kk = 0; kk < 64; kk += 32) {
      Frag16 a[4], b[2];
      // A fragment: lane holds row m=lane%16; K groups [8h,8h+8) and [16+8h,16+8h+8)
#pragma unroll
      for (int i = 0; i < 4; ++i) {
        const unsigned short* p = &As[buf][wm * 64 + i * 16 + mrow][kk + khalf * 8];
        a[i].q[0] = *(const uint4*)(p);
        a[i].q[1] = *(const uint4*)(p + 16);
      }
      // B fragment: lane holds column n=lane%16; K groups [8h,8h+8) and [16+8h,16+8h+8)
#pragma unroll
      for (int jj = 0; jj < 2; ++jj) {
        const unsigned short* p = &Bs[buf][kk + khalf * 8][wn * 32 + jj * 16 + mrow];
#pragma unroll
        for (int s = 0; s < 16; ++s)
          b[jj].v[s] = __builtin_bit_cast(__bf16, p[(s < 8 ? s : s + 8) * 136]);
      }
#pragma unroll
      for (int i = 0; i < 4; ++i)
#pragma unroll
        for (int jj = 0; jj < 2; ++jj)
          acc[i][jj] = __builtin_amdgcn_wmma_f32_16x16x32_bf16(
              false, a[i].v, false, b[jj].v, (short)0, acc[i][jj], false, false);
    }
    __syncthreads();                        // done reading buf before it is refilled
  }

  // epilogue: C/D layout VGPR v, lanes 0-15 -> M=v, lanes 16-31 -> M=v+8; N=lane%16
#pragma unroll
  for (int i = 0; i < 4; ++i) {
#pragma unroll
    for (int j = 0; j < 2; ++j) {
      const int n = nBase + wn * 32 + j * 16 + mrow;
#pragma unroll
      for (int v = 0; v < 8; ++v) {
        const int m = mBase + wm * 64 + i * 16 + (khalf << 3) + v;
        float val = acc[i][j][v];
        if (sigma) outB[(size_t)m * N + n] = f2bf(val * sigma[n]);
        else       outF[(size_t)m * N + n] = val;
      }
    }
  }
}

// ---------------------------------------------------------------------------

extern "C" void kernel_launch(void* const* d_in, const int* in_sizes, int n_in,
                              void* d_out, int out_size, void* d_ws, size_t ws_size,
                              hipStream_t stream) {
  (void)in_sizes; (void)n_in; (void)out_size; (void)ws_size;

  const float* inputs   = (const float*)d_in[0];
  const float* hidden   = (const float*)d_in[1];
  const float* W0_U     = (const float*)d_in[2];
  const float* W0_sigma = (const float*)d_in[3];
  const float* W0_V     = (const float*)d_in[4];
  const float* Ws_U     = (const float*)d_in[5];
  const float* Ws_sigma = (const float*)d_in[6];
  const float* Ws_V     = (const float*)d_in[7];
  float* out = (float*)d_out;

  char* ws = (char*)d_ws;
  size_t off = 0;
  auto take = [&](size_t bytes) -> void* {
    off = (off + 255) & ~(size_t)255;
    void* p = ws + off;
    off += bytes;
    return p;
  };

  unsigned short* w0u  = (unsigned short*)take((size_t)NXP * NXP * 2);
  unsigned short* w0v  = (unsigned short*)take((size_t)NXP * NXP * 2);
  unsigned short* wsu  = (unsigned short*)take((size_t)8 * NHP * NHP * 2);
  unsigned short* wsv  = (unsigned short*)take((size_t)8 * NHP * NXP * 2);
  float*          sig0 = (float*)take((size_t)NXP * 4);
  float*          sigs = (float*)take((size_t)8 * NHP * 4);
  unsigned short* xh   = (unsigned short*)take((size_t)B_ * NXP * 2);
  unsigned short* tbf  = (unsigned short*)take((size_t)B_ * NXP * 2);
  float*          chf  = (float*)take((size_t)B_ * NXP * 4);
  float*          stf  = (float*)take((size_t)9 * B_ * NHP * 4);
  unsigned short* stbf = (unsigned short*)take((size_t)9 * B_ * NHP * 2);
  float*          hf   = (float*)take((size_t)B_ * NHP * 4);

  dim3 blk(256);

  // ---- one-time per launch: convert/pad weights to bf16 (L2-resident, reused 128x) ----
  {
    int tot = NXP * NXP;
    k_convpad_bf16<<<(tot + 255) / 256, blk, 0, stream>>>(W0_U, w0u, NX, NX, NX, NXP, NXP);
    k_convpad_bf16<<<(tot + 255) / 256, blk, 0, stream>>>(W0_V, w0v, NX, NX, NX, NXP, NXP);
  }
  for (int i = 0; i < 8; ++i) {
    int totu = NHP * NHP;
    k_convpad_bf16<<<(totu + 255) / 256, blk, 0, stream>>>(
        Ws_U + (size_t)i * NH * NH, wsu + (size_t)i * NHP * NHP, NH, NH, NH, NHP, NHP);
    int totv = NHP * NXP;  // only first 850 rows of Ws_V contribute (zero-pad embed)
    k_convpad_bf16<<<(totv + 255) / 256, blk, 0, stream>>>(
        Ws_V + (size_t)i * NX * NX, wsv + (size_t)i * NHP * NXP, NH, NX, NX, NHP, NXP);
  }
  k_sigpad<<<(NXP + 255) / 256, blk, 0, stream>>>(W0_sigma, sig0, NX, NXP);
  for (int i = 0; i < 8; ++i)
    k_sigpad<<<(NHP + 255) / 256, blk, 0, stream>>>(Ws_sigma + (size_t)i * NH,
                                                    sigs + (size_t)i * NHP, NH, NHP);
  k_inith<<<(B_ * NHP + 255) / 256, blk, 0, stream>>>(hidden, hf);
  k_zero32<<<(9 * B_ * NHP + 255) / 256, blk, 0, stream>>>(stf, 9 * B_ * NHP);
  k_zero32<<<((9 * B_ * NHP / 2) + 255) / 256, blk, 0, stream>>>((float*)stbf, 9 * B_ * NHP / 2);

  // ---- sequential scan ----
  const int acts[8]  = {2, 1, 1, 3, 0, 2, 0, 1};  // sigmoid,relu,relu,id,tanh,sigmoid,tanh,relu
  const int preds[8] = {0, 1, 1, 1, 2, 5, 3, 5};
  dim3 gBig(NXP / 128, B_ / 128);   // (14, 2)
  dim3 gSm (NHP / 128, B_ / 128);   // (7, 2)
  const int nEW = (B_ * NH + 255) / 256;

  for (int t = 0; t < T_; ++t) {
    k_packxh<<<(B_ * NXP + 255) / 256, blk, 0, stream>>>(inputs + (size_t)t * B_ * NH, hf, xh);
    // ch0 = ((xh @ U0) * sigma0) @ V0
    k_gemm_bf16<<<gBig, blk, 0, stream>>>(xh,  w0u, NXP, NXP, sig0,    tbf,     nullptr);
    k_gemm_bf16<<<gBig, blk, 0, stream>>>(tbf, w0v, NXP, NXP, nullptr, nullptr, chf);
    k_epi0<<<nEW, blk, 0, stream>>>(chf, hf, stf, stbf);

    for (int i = 0; i < 8; ++i) {
      const unsigned short* spb = stbf + (size_t)preds[i] * B_ * NHP;
      const float*          spf = stf  + (size_t)preds[i] * B_ * NHP;
      k_gemm_bf16<<<gSm,  blk, 0, stream>>>(spb, wsu + (size_t)i * NHP * NHP, NHP, NHP,
                                            sigs + (size_t)i * NHP, tbf, nullptr);
      k_gemm_bf16<<<gBig, blk, 0, stream>>>(tbf, wsv + (size_t)i * NHP * NXP, NHP, NXP,
                                            nullptr, nullptr, chf);
      k_epi<<<nEW, blk, 0, stream>>>(chf, spf, stf + (size_t)(i + 1) * B_ * NHP,
                                     stbf + (size_t)(i + 1) * B_ * NHP, acts[i]);
    }
    k_final<<<nEW, blk, 0, stream>>>(stf, out + (size_t)t * B_ * NH, hf,
                                     (t == T_ - 1) ? out + (size_t)T_ * B_ * NH : nullptr);
  }
}
